// FolkRNN_78718160601402
// MI455X (gfx1250) — matrix operand order repaired and checked by reading, not compiled, not run on hardware
//
#include <hip/hip_runtime.h>
#include <hip/hip_bf16.h>
#include <stdint.h>

// ---------------- problem constants ----------------
#define VV   2048
#define EE   512
#define HH   512
#define LL   3
#define TT   512
#define BB   64
#define TN   (TT - 1)          // 511 steps
#define MROWS (TN * BB)        // 32704
#define MT   (MROWS / 16)      // 2044 m-tiles
#define KT16 (EE / 32)         // 16 k-tiles (K = 512)
#define G4H  (4 * HH)          // 2048
#define NT_G (G4H / 16)        // 128 n-tiles
#define NWG  16                // recurrence workgroups (each owns 32 h-cols)

typedef __attribute__((ext_vector_type(16))) __bf16 v16bf;
typedef __attribute__((ext_vector_type(8)))  float  v8f;
typedef __attribute__((ext_vector_type(4)))  unsigned int u32x4;
typedef __attribute__((ext_vector_type(4)))  int   i32x4;
typedef __attribute__((ext_vector_type(8)))  int   i32x8;

#if defined(__HIP_DEVICE_COMPILE__)
# if __has_builtin(__builtin_amdgcn_tensor_load_to_lds)
#  define HAVE_TDM 1
# endif
# if __has_builtin(__builtin_amdgcn_s_wait_tensorcnt)
#  define HAVE_TWAIT 1
# endif
#endif

// A-matrix (16x32 bf16) per-lane element -> K offset within the 32-wide k-tile
// (ISA 7.12.2, 16-bit A table)
__host__ __device__ inline int a_frag_k(int lane, int e) {
    int v = e >> 1;
    return ((v & 3) << 1) | (e & 1) | ((v & 4) << 2) | ((lane >> 4) << 3);
}
// B-matrix (32x16 bf16): same K order as the A table (lanes 16-31 hold K+8)
__host__ __device__ inline int b_frag_k(int lane, int e) {
    return a_frag_k(lane, e);
}

__device__ inline v8f wmma_bf16(v16bf a, v16bf b, v8f c) {
#if defined(__HIP_DEVICE_COMPILE__)
    return __builtin_amdgcn_wmma_f32_16x16x32_bf16(false, a, false, b, (short)0, c,
                                                   false, false);
#else
    return c;
#endif
}

__device__ inline float sigmoidf_fast(float x) {
#if defined(__HIP_DEVICE_COMPILE__)
    return 1.0f / (1.0f + __expf(-x));
#else
    return x;
#endif
}

// ---------------- weight pack: fp32 (N x K) row-major -> bf16 B-fragments ----
__global__ void pack_w_bfrag(const float* __restrict__ W, __bf16* __restrict__ out,
                             int N, int K) {
    int idx = blockIdx.x * blockDim.x + threadIdx.x;   // one per (tile, lane)
    int KT = K / 32;
    int total = (N / 16) * KT * 32;
    if (idx >= total) return;
    int lane = idx & 31;
    int rest = idx >> 5;
    int kt = rest % KT;
    int nt = rest / KT;
    int n = nt * 16 + (lane & 15);
    __bf16* dst = out + (size_t)idx * 16;
    const float* src = W + (size_t)n * K + kt * 32;
    for (int e = 0; e < 16; ++e)
        dst[e] = (__bf16)src[b_frag_k(lane, e)];
}

// ---------------- embedding gather -> bf16 A-fragments ----------------------
__global__ void embed_afrag(const int* __restrict__ tokens,
                            const float* __restrict__ emb,
                            __bf16* __restrict__ out) {
    int idx = blockIdx.x * blockDim.x + threadIdx.x;   // (mt*16 + kt)*32 + lane
    int total = MT * KT16 * 32;
    if (idx >= total) return;
    int lane = idx & 31;
    int rest = idx >> 5;
    int kt = rest & 15;
    int mt = rest >> 4;
    int m = mt * 16 + (lane & 15);                     // m = t*64 + b
    int tok = tokens[m];
    __bf16* dst = out + (size_t)idx * 16;
    const float* src = emb + (size_t)tok * EE + kt * 32;
    for (int e = 0; e < 16; ++e)
        dst[e] = (__bf16)src[a_frag_k(lane, e)];
}

// ---------------- tiled WMMA GEMM: Afrag (MxK) * Bfrag (KxN) + bias -> f32 ---
// block = 256 threads (8 waves, 4m x 2n). Block tile: 256M x 64N.
// Wave tile: 64M x 32N = 8 accum tiles -> 8 WMMAs per 12 b128 loads.
__global__ void gemm_frag_bias(const __bf16* __restrict__ Afrag,
                               const __bf16* __restrict__ Bfrag,
                               const float* __restrict__ bias,
                               float* __restrict__ out,
                               int Mtiles, int Ntiles, int KT, int Nld) {
    const int lane = threadIdx.x & 31;
    const int wave = threadIdx.x >> 5;
    const int wm = wave >> 1;                  // 0..3
    const int wn = wave & 1;                   // 0..1
    const int mt0 = blockIdx.y * 16 + wm * 4;  // 4 m-tiles
    const int nt0 = blockIdx.x * 4 + wn * 2;   // 2 n-tiles
    if (nt0 >= Ntiles) return;

    v8f acc[4][2] = {};
    bool mok[4];
    int mti[4];
    for (int i = 0; i < 4; ++i) {
        mok[i] = (mt0 + i) < Mtiles;
        mti[i] = mok[i] ? (mt0 + i) : (Mtiles - 1);   // clamp: harmless dup load
    }

    for (int kt = 0; kt < KT; ++kt) {
        v16bf a[4], b[2];
        for (int i = 0; i < 4; ++i)
            a[i] = *(const v16bf*)(Afrag + ((size_t)(mti[i] * KT + kt) * 32 + lane) * 16);
        for (int j = 0; j < 2; ++j)
            b[j] = *(const v16bf*)(Bfrag + ((size_t)((nt0 + j) * KT + kt) * 32 + lane) * 16);
        if (kt + 1 < KT)  // hint the B stream (global_prefetch_b8)
            __builtin_prefetch(Bfrag + ((size_t)(nt0 * KT + kt + 1) * 32 + lane) * 16, 0, 1);
        for (int i = 0; i < 4; ++i)
            for (int j = 0; j < 2; ++j)
                acc[i][j] = wmma_bf16(a[i], b[j], acc[i][j]);
    }

    // C/D layout: lane -> N = lane&15, VGPR r -> M = r + 8*(lane>=16)
    for (int i = 0; i < 4; ++i) {
        if (!mok[i]) continue;
        int mrow = (mt0 + i) * 16 + (lane >> 4) * 8;
        for (int j = 0; j < 2; ++j) {
            int n = (nt0 + j) * 16 + (lane & 15);
            float bv = bias ? bias[n] : 0.0f;
            for (int r = 0; r < 8; ++r)
                out[(size_t)(mrow + r) * Nld + n] = acc[i][j][r] + bv;
        }
    }
}

// ---------------- persistent LSTM recurrence --------------------------------
// 16 workgroups x 256 threads. Workgroup wg owns h columns [wg*32, wg*32+32),
// i.e. exactly k-tile `wg` of the next step's A operand. Its 8 gate n-tiles
// (2 per gate block) are computed by 8 waves (1 n-tile each, 4 m-tiles deep).
// The workgroup's W_hh slice (8 nt x 16 kt = 128 KB bf16) is staged into LDS
// ONCE via the Tensor Data Mover, then all 511 steps read B operands from LDS.
__global__ void lstm_recurrence(const float* __restrict__ xg,     // (TN*64) x 2048
                                const __bf16* __restrict__ WhhF,  // B-frags 128nt x 16kt
                                const float* __restrict__ bhh,    // 2048
                                __bf16* __restrict__ ysF,         // A-frags (TN*4 mt x 16 kt)
                                unsigned* __restrict__ bar) {
    extern __shared__ char smem_raw[];
    __bf16* whh_lds = (__bf16*)smem_raw;                    // 128 KB (65536 bf16)
    float*  gbuf    = (float*)(smem_raw + 128 * 1024);      // 32 KB (4 x 64 x 32 f32)

    const int wg   = blockIdx.x;                            // 0..15
    const int tid  = threadIdx.x;
    const int lane = tid & 31;
    const int wave = tid >> 5;                              // 0..7
    const int Gg   = wave >> 1;                             // gate block 0..3
    const int j    = wave & 1;                              // local n-tile 0..1
    const int nt_g = Gg * 32 + wg * 2 + j;                  // global gate n-tile
    const int KT   = KT16;

    // ---- stage this workgroup's W_hh slice into LDS (once) ----
    {
        const __bf16* src = WhhF + (size_t)nt_g * KT * 32 * 16;   // 16 KB contiguous
#if defined(HAVE_TDM)
        // One TDM descriptor per wave: copy 16 KB (8192 bf16) global -> LDS.
        uint64_t gaddr = (uint64_t)(uintptr_t)src;
        unsigned ldsoff = (unsigned)(wave * 16384);               // bytes
        u32x4 g0;
        g0[0] = 1u;                                               // count=1, user mode
        g0[1] = ldsoff;                                           // lds_addr
        g0[2] = (unsigned)gaddr;                                  // global_addr[31:0]
        g0[3] = ((unsigned)(gaddr >> 32) & 0x01FFFFFFu) | (2u << 30); // addr[56:32], type=2
        i32x8 g1;
        g1[0] = (1 << 16);                // workgroup_mask=0, data_size=1 (2B)
        g1[1] = (0x2000 << 16);           // tensor_dim0 = 8192 (low 16 bits)
        g1[2] = (1 << 16);                // tensor_dim0 hi=0, tensor_dim1 = 1
        g1[3] = (0x2000 << 16);           // tile_dim0 = 8192
        g1[4] = 1;                        // tile_dim1 = 1, tile_dim2 = 0
        g1[5] = 0x2000;                   // tensor_dim0_stride = 8192
        g1[6] = 0;
        g1[7] = 0;
        i32x4 z4 = {0, 0, 0, 0};
# if __clang_major__ >= 23
        i32x8 z8 = {0, 0, 0, 0, 0, 0, 0, 0};
        __builtin_amdgcn_tensor_load_to_lds(g0, g1, z4, z4, z8, 0);
# else
        __builtin_amdgcn_tensor_load_to_lds(g0, g1, z4, z4, 0);
# endif
#else
        // fallback: plain global->LDS copy (512 x 32B chunks per wave)
        v16bf* ldst = (v16bf*)whh_lds + (size_t)wave * 512;
        const v16bf* gsrc = (const v16bf*)src;
        for (int it = lane; it < 512; it += 32)
            ldst[it] = gsrc[it];
#endif
    }
#if defined(HAVE_TWAIT)
    __builtin_amdgcn_s_wait_tensorcnt(0);
#endif
    __syncthreads();

    float c[8];
    for (int q = 0; q < 8; ++q) c[q] = 0.0f;
    unsigned phase = 0;

    for (int t = 0; t < TN; ++t) {
        // 1) acc = h_{t-1} @ Whh^T : A from global (prev step's frags), B from LDS
        v8f acc[4] = {};
        if (t > 0) {
            const __bf16* hF = ysF + (size_t)(t - 1) * 4 * KT * 32 * 16;
            for (int kt = 0; kt < KT; ++kt) {
                const v16bf b =
                    *(const v16bf*)(whh_lds + ((size_t)(wave * KT + kt) * 32 + lane) * 16);
                for (int mt = 0; mt < 4; ++mt) {
                    v16bf a = *(const v16bf*)(hF + ((size_t)(mt * KT + kt) * 32 + lane) * 16);
                    acc[mt] = wmma_bf16(a, b, acc[mt]);
                }
            }
        }

        // 2) pre-activations -> LDS: gbuf[(G*64 + b)*32 + hc_local]
        {
            int n = Gg * 512 + wg * 32 + j * 16 + (lane & 15);
            float bb = bhh[n];
            for (int mt = 0; mt < 4; ++mt) {
                int brow = mt * 16 + (lane >> 4) * 8;
                for (int r = 0; r < 8; ++r) {
                    int b_ = brow + r;
                    float g = acc[mt][r] + xg[((size_t)t * 64 + b_) * (size_t)G4H + n] + bb;
                    gbuf[(size_t)(Gg * 64 + b_) * 32 + (j * 16 + (lane & 15))] = g;
                }
            }
        }
        __syncthreads();

        // 3) elementwise cell update; each thread owns 8 (b, hc) elements
        for (int q = 0; q < 8; ++q) {
            int flat = tid * 8 + q;
            int b_ = flat >> 5, hc = flat & 31;
            float i_ = gbuf[(0 * 64 + b_) * 32 + hc];
            float f_ = gbuf[(1 * 64 + b_) * 32 + hc];
            float g_ = gbuf[(2 * 64 + b_) * 32 + hc];
            float o_ = gbuf[(3 * 64 + b_) * 32 + hc];
            i_ = sigmoidf_fast(i_);
            f_ = sigmoidf_fast(f_);
            g_ = tanhf(g_);
            o_ = sigmoidf_fast(o_);
            float cn = f_ * c[q] + i_ * g_;
            c[q] = cn;
            gbuf[(size_t)b_ * 32 + hc] = o_ * tanhf(cn);   // only owner touches it
        }
        __syncthreads();

        // 4) write h_t into ys A-fragments: this wg owns k-tile kt = wg
        {
            __bf16* dst = ysF + (size_t)t * 4 * KT * 32 * 16;
            for (int q = 0; q < 8; ++q) {
                int flat = tid * 8 + q;              // (mt*32 + ln)*16 + e
                int e = flat & 15;
                int rest = flat >> 4;
                int ln = rest & 31;
                int mt = rest >> 5;                  // 0..3
                int hc = a_frag_k(ln, e);            // column within this k-tile
                int b_ = mt * 16 + (ln & 15);
                dst[((size_t)(mt * KT + wg) * 32 + ln) * 16 + e] =
                    (__bf16)gbuf[(size_t)b_ * 32 + hc];
            }
        }

        // 5) grid barrier (cumulative counters; bar[] zeroed per launch)
        __threadfence();
        __syncthreads();
        if (tid == 0) {
            unsigned arrived = atomicAdd(&bar[0], 1u) + 1u;
            if (arrived == (phase + 1u) * gridDim.x) atomicAdd(&bar[1], 1u);
            while (atomicAdd(&bar[1], 0u) <= phase) {
#if defined(__HIP_DEVICE_COMPILE__)
                __builtin_amdgcn_s_sleep(1);
#endif
            }
        }
        __syncthreads();
        __threadfence();
        ++phase;
    }
}

// ---------------- argmax sampler (stand-in for categorical) -----------------
__global__ void argmax_rows(const float* __restrict__ logits, float* __restrict__ outv,
                            int N) {
    __shared__ float sv[256];
    __shared__ int   si[256];
    int row = blockIdx.x;
    const float* p = logits + (size_t)row * N;
    float best = -3.4e38f;
    int bi = 0;
    for (int n = threadIdx.x; n < N; n += 256) {
        float v = p[n];
        if (v > best) { best = v; bi = n; }
    }
    sv[threadIdx.x] = best;
    si[threadIdx.x] = bi;
    __syncthreads();
    for (int s = 128; s > 0; s >>= 1) {
        if (threadIdx.x < s && sv[threadIdx.x + s] > sv[threadIdx.x]) {
            sv[threadIdx.x] = sv[threadIdx.x + s];
            si[threadIdx.x] = si[threadIdx.x + s];
        }
        __syncthreads();
    }
    if (threadIdx.x == 0) outv[row] = (float)si[0];
}

// ---------------- host orchestration ----------------------------------------
extern "C" void kernel_launch(void* const* d_in, const int* in_sizes, int n_in,
                              void* d_out, int out_size, void* d_ws, size_t ws_size,
                              hipStream_t stream) {
    const int*   tokens = (const int*)  d_in[0];   // (512, 64) int32
    const float* emb    = (const float*)d_in[1];   // (2048, 512)
    const float* Wih    = (const float*)d_in[2];   // (3, 2048, 512)
    const float* Whh    = (const float*)d_in[3];   // (3, 2048, 512)
    const float* bih    = (const float*)d_in[4];   // (3, 2048)
    const float* bhh    = (const float*)d_in[5];   // (3, 2048)
    const float* Wout   = (const float*)d_in[6];   // (2048, 512)
    const float* bout   = (const float*)d_in[7];   // (2048)
    float* out = (float*)d_out;                    // [tokens 32704][logits 32704*2048]

    char* ws = (char*)d_ws;
    const size_t FRAG_BYTES_X = (size_t)MT * KT16 * 32 * 16 * sizeof(__bf16);   // 33.5 MB
    const size_t FRAG_BYTES_W = (size_t)NT_G * KT16 * 32 * 16 * sizeof(__bf16); // 2 MB
    size_t o_bar  = 0;
    size_t o_xA   = 256;
    size_t o_xB   = o_xA + FRAG_BYTES_X;
    size_t o_wih  = o_xB + FRAG_BYTES_X;
    size_t o_whh  = o_wih + 3 * FRAG_BYTES_W;
    size_t o_wout = o_whh + 3 * FRAG_BYTES_W;
    size_t o_xg   = o_wout + FRAG_BYTES_W;
    // total ~ 350 MB of workspace

    unsigned* bar   = (unsigned*)(ws + o_bar);
    __bf16* xA      = (__bf16*)(ws + o_xA);
    __bf16* xB      = (__bf16*)(ws + o_xB);
    __bf16* wihF    = (__bf16*)(ws + o_wih);
    __bf16* whhF    = (__bf16*)(ws + o_whh);
    __bf16* woutF   = (__bf16*)(ws + o_wout);
    float*  xg      = (float*) (ws + o_xg);

    const size_t WFRAG_ELEMS = (size_t)NT_G * KT16 * 32 * 16;

    // pack weights -> bf16 fragments
    {
        int total = NT_G * KT16 * 32;                  // 65536 threads
        int blocks = (total + 255) / 256;
        for (int l = 0; l < LL; ++l) {
            pack_w_bfrag<<<blocks, 256, 0, stream>>>(Wih + (size_t)l * G4H * EE,
                                                     wihF + (size_t)l * WFRAG_ELEMS,
                                                     G4H, EE);
            pack_w_bfrag<<<blocks, 256, 0, stream>>>(Whh + (size_t)l * G4H * HH,
                                                     whhF + (size_t)l * WFRAG_ELEMS,
                                                     G4H, HH);
        }
        pack_w_bfrag<<<blocks, 256, 0, stream>>>(Wout, woutF, VV, HH);
    }

    // embedding gather -> A-fragments
    {
        int total = MT * KT16 * 32;
        embed_afrag<<<(total + 255) / 256, 256, 0, stream>>>(tokens, emb, xA);
    }

    __bf16* cur = xA;
    __bf16* nxt = xB;
    dim3 ggrid(NT_G / 4, (MT + 15) / 16);              // (32, 128)
    for (int l = 0; l < LL; ++l) {
        // xg = x @ W_ih^T + b_ih   (bulk WMMA GEMM)
        gemm_frag_bias<<<ggrid, 256, 0, stream>>>(cur, wihF + (size_t)l * WFRAG_ELEMS,
                                                  bih + (size_t)l * G4H, xg,
                                                  MT, NT_G, KT16, G4H);
        // sequential recurrence (persistent, LDS-resident W_hh, grid barrier/step)
        hipMemsetAsync(bar, 0, 256, stream);
        lstm_recurrence<<<NWG, 256, 160 * 1024, stream>>>(
            xg, whhF + (size_t)l * WFRAG_ELEMS, bhh + (size_t)l * G4H, nxt, bar);
        __bf16* tmp = cur; cur = nxt; nxt = tmp;
    }

    // logits = ys @ W_out^T + b_out  -> d_out[32704 ...]
    float* logits = out + MROWS;
    gemm_frag_bias<<<ggrid, 256, 0, stream>>>(cur, woutF, bout, logits,
                                              MT, NT_G, KT16, VV);

    // "sampled" tokens (argmax stand-in; reference PRNG not reproducible)
    argmax_rows<<<MROWS, 256, 0, stream>>>(logits, out, VV);
}
